// GraphModule_30812095381601
// MI455X (gfx1250) — matrix operand order repaired and checked by reading, not compiled, not run on hardware
//
#include <hip/hip_runtime.h>

// ---------------------------------------------------------------------------
// Types for CDNA5 WMMA (wave32, 16x16x32 bf16 -> f32)
// ---------------------------------------------------------------------------
typedef __attribute__((ext_vector_type(16))) __bf16 v16bf;
typedef __attribute__((ext_vector_type(8)))  float  v8f;

union Frag32B {           // one WMMA A/B operand: 16 bf16 = 32 bytes = 8 VGPRs
    v16bf v;
    uint4 u[2];
};

static __device__ __forceinline__ unsigned short f2bf_rne(float f) {
    unsigned u = __float_as_uint(f);
    u += 0x7FFFu + ((u >> 16) & 1u);        // round-to-nearest-even to bf16
    return (unsigned short)(u >> 16);
}
static __device__ __forceinline__ float bf2f(unsigned short h) {
    return __uint_as_float(((unsigned)h) << 16);
}

// ---------------------------------------------------------------------------
// Elementwise / graph-norm kernels
// ---------------------------------------------------------------------------
__global__ void k_init_deg(float* __restrict__ deg, int n) {
    int i = blockIdx.x * blockDim.x + threadIdx.x;
    if (i < n) deg[i] = 1.0f;               // self-loop weight
}

__global__ void k_deg_accum(const int* __restrict__ ei, const float* __restrict__ ew,
                            float* __restrict__ deg, int nE) {
    int e = blockIdx.x * blockDim.x + threadIdx.x;
    if (e < nE) atomicAdd(&deg[ei[nE + e]], ew[e]);   // dst = row 1
}

__global__ void k_rsqrt(float* __restrict__ d, int n) {
    int i = blockIdx.x * blockDim.x + threadIdx.x;
    if (i < n) { float v = d[i]; d[i] = (v > 0.f) ? rsqrtf(v) : 0.f; }
}

// split fp32 -> bf16 hi + bf16 lo residual (bf16x3 GEMM inputs)
__global__ void k_split_bf16(const float* __restrict__ in, unsigned short* __restrict__ hi,
                             unsigned short* __restrict__ lo, int n) {
    int i = blockIdx.x * blockDim.x + threadIdx.x;
    if (i >= n) return;
    float x = in[i];
    unsigned short h = f2bf_rne(x);
    hi[i] = h;
    lo[i] = f2bf_rne(x - bf2f(h));
}

__global__ void k_init_bias(float* __restrict__ out, const float* __restrict__ bias,
                            int total, int dmask) {
    int i = blockIdx.x * blockDim.x + threadIdx.x;
    if (i < total) out[i] = bias[i & dmask];          // D is a power of two
}

__global__ void k_relu(float* __restrict__ x, int n) {
    int i = blockIdx.x * blockDim.x + threadIdx.x;
    if (i < n) x[i] = fmaxf(x[i], 0.f);
}

// ---------------------------------------------------------------------------
// Edge aggregation: OUT[dst] += norm * H[src]  (blockDim.x == D, coalesced)
// blocks [0,E) are real edges, blocks [E, E+N) are self-loops.
// ---------------------------------------------------------------------------
__global__ void k_aggregate(const float* __restrict__ H, const int* __restrict__ ei,
                            const float* __restrict__ ew, const float* __restrict__ dinv,
                            float* __restrict__ OUT, int nE) {
    const int e = blockIdx.x;
    const int c = threadIdx.x;
    const int D = blockDim.x;
    int s, d; float nrm;
    if (e < nE) {
        s = ei[e]; d = ei[nE + e];
        nrm = dinv[s] * ew[e] * dinv[d];
    } else {
        s = d = e - nE;
        float dv = dinv[s];
        nrm = dv * dv;
    }
    float val = H[(size_t)s * D + c] * nrm;
    atomicAdd(&OUT[(size_t)d * D + c], val);
}

// ---------------------------------------------------------------------------
// bf16x3 WMMA GEMM: C[M x NC] = A[M x K] @ W[NC x K]^T   (fp32 accumulate)
// One wave -> one 16x16 tile. block = (32, 8); grid = (M/16, NC/128).
// ---------------------------------------------------------------------------
__global__ __launch_bounds__(256) void k_gemm_bf16x3(
    const unsigned short* __restrict__ Ah, const unsigned short* __restrict__ Al,
    const unsigned short* __restrict__ Wh, const unsigned short* __restrict__ Wl,
    float* __restrict__ C, int M, int K, int NC)
{
    const int lane = threadIdx.x;                       // 0..31
    const int row0 = blockIdx.x * 16;
    const int col0 = (blockIdx.y * 8 + threadIdx.y) * 16;
    if (row0 >= M || col0 >= NC) return;                // wave-uniform, EXEC stays full

    const int half = lane >> 4;                         // K sub-block select
    const int m    = row0 + (lane & 15);                // A row for this lane
    const int n    = col0 + (lane & 15);                // B column for this lane

    const unsigned short* aRowH = Ah + (size_t)m * K;
    const unsigned short* aRowL = Al + (size_t)m * K;
    const unsigned short* wRowH = Wh + (size_t)n * K;
    const unsigned short* wRowL = Wl + (size_t)n * K;

    v8f acc = {0.f, 0.f, 0.f, 0.f, 0.f, 0.f, 0.f, 0.f};

    for (int kb = 0; kb < K; kb += 32) {
        Frag32B fah, fal, fbh, fbl;
        // A 16x32 bf16 layout: lane half h, elems 0-7 -> K = kb+8h+j,
        //                                  elems 8-15 -> K = kb+16+8h+j
        const unsigned short* pah = aRowH + kb + half * 8;
        fah.u[0] = *(const uint4*)(pah);
        fah.u[1] = *(const uint4*)(pah + 16);
        const unsigned short* pal = aRowL + kb + half * 8;
        fal.u[0] = *(const uint4*)(pal);
        fal.u[1] = *(const uint4*)(pal + 16);
        // B 32x16 bf16 layout: lane half h, elems 0-7 -> K = kb+8h+j,
        //                                  elems 8-15 -> K = kb+16+8h+j
        const unsigned short* pbh = wRowH + kb + half * 8;
        fbh.u[0] = *(const uint4*)(pbh);
        fbh.u[1] = *(const uint4*)(pbh + 16);
        const unsigned short* pbl = wRowL + kb + half * 8;
        fbl.u[0] = *(const uint4*)(pbl);
        fbl.u[1] = *(const uint4*)(pbl + 16);

        // bf16x3: x*w ~= xh*wh + xh*wl + xl*wh
        acc = __builtin_amdgcn_wmma_f32_16x16x32_bf16(false, fah.v, false, fbh.v,
                                                      (short)0, acc, false, false);
        acc = __builtin_amdgcn_wmma_f32_16x16x32_bf16(false, fah.v, false, fbl.v,
                                                      (short)0, acc, false, false);
        acc = __builtin_amdgcn_wmma_f32_16x16x32_bf16(false, fal.v, false, fbh.v,
                                                      (short)0, acc, false, false);
    }

    // C/D layout: VGPR v -> row row0 + v + 8*(lane>=16); col = col0 + (lane&15)
    float* cBase = C + (size_t)(row0 + half * 8) * NC + n;
#pragma unroll
    for (int v = 0; v < 8; ++v)
        cBase[(size_t)v * NC] = acc[v];
}

// ---------------------------------------------------------------------------
// LayerNorm over D=128 (one wave per node; 4 features/lane; shfl reduction)
// ---------------------------------------------------------------------------
__global__ __launch_bounds__(256) void k_layernorm128(
    const float* __restrict__ H, const float* __restrict__ gamma,
    const float* __restrict__ beta, float* __restrict__ out, int nN)
{
    const int lane = threadIdx.x & 31;
    const int node = blockIdx.x * 8 + (threadIdx.x >> 5);
    if (node >= nN) return;
    const float* row = H + (size_t)node * 128;
    float v0 = row[lane], v1 = row[lane + 32], v2 = row[lane + 64], v3 = row[lane + 96];
    float s = v0 + v1 + v2 + v3;
#pragma unroll
    for (int off = 16; off; off >>= 1) s += __shfl_xor(s, off, 32);
    float mu = s * (1.0f / 128.0f);
    float d0 = v0 - mu, d1 = v1 - mu, d2 = v2 - mu, d3 = v3 - mu;
    float q = d0 * d0 + d1 * d1 + d2 * d2 + d3 * d3;
#pragma unroll
    for (int off = 16; off; off >>= 1) q += __shfl_xor(q, off, 32);
    float rstd = rsqrtf(q * (1.0f / 128.0f) + 1e-5f);
    float* o = out + (size_t)node * 128;
    o[lane]      = d0 * rstd * gamma[lane]      + beta[lane];
    o[lane + 32] = d1 * rstd * gamma[lane + 32] + beta[lane + 32];
    o[lane + 64] = d2 * rstd * gamma[lane + 64] + beta[lane + 64];
    o[lane + 96] = d3 * rstd * gamma[lane + 96] + beta[lane + 96];
}

// ---------------------------------------------------------------------------
// Host orchestration
// ---------------------------------------------------------------------------
extern "C" void kernel_launch(void* const* d_in, const int* in_sizes, int n_in,
                              void* d_out, int out_size, void* d_ws, size_t ws_size,
                              hipStream_t stream) {
    const float* x     = (const float*)d_in[0];
    const int*   ei    = (const int*)d_in[1];      // [2, E] flat: row0=src, row1=dst
    const float* ew    = (const float*)d_in[2];
    const float* W1    = (const float*)d_in[3];    // [DHID, DIN]
    const float* b1    = (const float*)d_in[4];
    const float* W2    = (const float*)d_in[5];    // [DIN, DHID]
    const float* b2    = (const float*)d_in[6];
    const float* gamma = (const float*)d_in[7];
    const float* beta  = (const float*)d_in[8];
    float* out = (float*)d_out;

    const int DIN  = 128;
    const int DHID = in_sizes[4];                  // 256
    const int E    = in_sizes[2];                  // 800000
    const int N    = in_sizes[0] / DIN;            // 50000

    // ---- workspace layout (256B aligned regions, with reuse) ----
    char* ws = (char*)d_ws;
    size_t cur = 0;
    auto take = [&cur](size_t bytes) {
        size_t o = cur;
        cur = (cur + bytes + 255) & ~(size_t)255;
        return o;
    };
    const size_t oDINV = take((size_t)N * 4);
    const size_t oW1H  = take((size_t)DHID * DIN * 2);
    const size_t oW1L  = take((size_t)DHID * DIN * 2);
    const size_t oW2H  = take((size_t)DIN * DHID * 2);
    const size_t oW2L  = take((size_t)DIN * DHID * 2);
    const size_t oXH   = take((size_t)N * DIN * 2);
    const size_t oXL   = take((size_t)N * DIN * 2);
    const size_t oH1   = take((size_t)N * DHID * 4);
    const size_t oA1   = take((size_t)N * DHID * 4);
    // reuse of dead regions (DHID == 2*DIN makes these exact fits):
    const size_t oA1H  = oXH;                        // N*DHID*2 over XH+XL
    const size_t oA1L  = oH1;                        // first half of H1 region
    const size_t oH2   = oH1 + (size_t)N * DHID * 2; // second half of H1 region
    const size_t oAGG2 = oA1;                        // first half of A1 region

    float*          dinv = (float*)(ws + oDINV);
    unsigned short* w1h  = (unsigned short*)(ws + oW1H);
    unsigned short* w1l  = (unsigned short*)(ws + oW1L);
    unsigned short* w2h  = (unsigned short*)(ws + oW2H);
    unsigned short* w2l  = (unsigned short*)(ws + oW2L);
    unsigned short* xh   = (unsigned short*)(ws + oXH);
    unsigned short* xl   = (unsigned short*)(ws + oXL);
    float*          h1   = (float*)(ws + oH1);
    float*          a1   = (float*)(ws + oA1);
    unsigned short* a1h  = (unsigned short*)(ws + oA1H);
    unsigned short* a1l  = (unsigned short*)(ws + oA1L);
    float*          h2   = (float*)(ws + oH2);
    float*          agg2 = (float*)(ws + oAGG2);

    auto cdiv = [](int a, int b) { return (a + b - 1) / b; };

    // 1) gcn_norm: deg (init self-loop=1) -> atomic accumulate -> rsqrt
    k_init_deg<<<cdiv(N, 256), 256, 0, stream>>>(dinv, N);
    k_deg_accum<<<cdiv(E, 256), 256, 0, stream>>>(ei, ew, dinv, E);
    k_rsqrt<<<cdiv(N, 256), 256, 0, stream>>>(dinv, N);

    // 2) bf16 hi/lo splits of X, W1, W2
    k_split_bf16<<<cdiv(N * DIN, 256), 256, 0, stream>>>(x, xh, xl, N * DIN);
    k_split_bf16<<<cdiv(DHID * DIN, 256), 256, 0, stream>>>(W1, w1h, w1l, DHID * DIN);
    k_split_bf16<<<cdiv(DIN * DHID, 256), 256, 0, stream>>>(W2, w2h, w2l, DIN * DHID);

    // 3) H1 = X @ W1^T  (WMMA bf16x3)
    k_gemm_bf16x3<<<dim3(cdiv(N, 16), DHID / 128), dim3(32, 8), 0, stream>>>(
        xh, xl, w1h, w1l, h1, N, DIN, DHID);

    // 4) A1 = segment_sum(norm * H1[src]) + b1 ; ReLU
    k_init_bias<<<cdiv(N * DHID, 256), 256, 0, stream>>>(a1, b1, N * DHID, DHID - 1);
    k_aggregate<<<E + N, DHID, 0, stream>>>(h1, ei, ew, dinv, a1, E);
    k_relu<<<cdiv(N * DHID, 256), 256, 0, stream>>>(a1, N * DHID);

    // 5) H2 = A1 @ W2^T  (WMMA bf16x3)
    k_split_bf16<<<cdiv(N * DHID, 256), 256, 0, stream>>>(a1, a1h, a1l, N * DHID);
    k_gemm_bf16x3<<<dim3(cdiv(N, 16), DIN / 128), dim3(32, 8), 0, stream>>>(
        a1h, a1l, w2h, w2l, h2, N, DHID, DIN);

    // 6) AGG2 = segment_sum(norm * H2[src]) + b2
    k_init_bias<<<cdiv(N * DIN, 256), 256, 0, stream>>>(agg2, b2, N * DIN, DIN - 1);
    k_aggregate<<<E + N, DIN, 0, stream>>>(h2, ei, ew, dinv, agg2, E);

    // 7) LayerNorm -> d_out
    k_layernorm128<<<cdiv(N, 8), 256, 0, stream>>>(agg2, gamma, beta, out, N);
}
